// GVPEncoder_33535104647909
// MI455X (gfx1250) — matrix operand order repaired and checked by reading, not compiled, not run on hardware
//
#include <hip/hip_runtime.h>
#include <math.h>

// ---------------------------------------------------------------------------
// CDNA5 (gfx1250) GVP-GNN encoder.
// All GEMMs run on v_wmma_f32_16x16x32_f16 (f16 operands, f32 accumulation).
// Weights are pre-swizzled into per-lane fragment layout so each B fragment
// is two global_load_b128; LDS A fragments are two ds_load_b128. K-chains are
// split into two independent even/odd accumulators (branch-free) so the
// scheduler can alternate WMMA streams and hide the f16 WMMA->WMMA RAW
// hazard without v_nops or exec-masked WMMA.
// ---------------------------------------------------------------------------

typedef __attribute__((ext_vector_type(16))) _Float16 v16h;
typedef __attribute__((ext_vector_type(8)))  _Float16 v8h;
typedef __attribute__((ext_vector_type(4)))  _Float16 v4h;
typedef __attribute__((ext_vector_type(8)))  float    v8f;

#define NWAVE 4
#define TB 128

struct GvpWB {
  const _Float16* wh;  const _Float16* ws;  const _Float16* wv;  const _Float16* wsv;
  const float* ws_b;   const float* wsv_b;
};
struct MsgW { GvpWB g[3]; };
struct FfW  { GvpWB g[2]; };

// ---- WMMA fragment helpers (layouts per CDNA5 ISA 7.12.2, wave32) ----------

// A: 16-bit 16x32 tile from LDS, row-major. Lane reads halves
// [base..base+7] and [base+16..base+23]; both 16B aligned.
__device__ __forceinline__ v16h frag_a(const _Float16* A, int lda, int m0, int k0, int lane) {
  const _Float16* p = A + (size_t)(m0 + (lane & 15)) * lda + k0 + (((lane >> 4) & 1) << 3);
  v8h lo = *(const v8h*)p;
  v8h hi = *(const v8h*)(p + 16);
  return __builtin_shufflevector(lo, hi, 0, 1, 2, 3, 4, 5, 6, 7,
                                 8, 9, 10, 11, 12, 13, 14, 15);
}

// B: pre-swizzled weights. Fragment for (kt, nt, lane) is 16 contiguous
// halves: elem e -> K = kt*32 + (lane>>4)*8 + (e&7) + (e>>3)*16, N = nt*16 + (lane&15).
__device__ __forceinline__ v16h frag_bs(const _Float16* B, int Ct, int kt, int nt, int lane) {
  return *(const v16h*)(B + ((((size_t)kt * Ct + nt) * 32) + lane) * 16);
}

__device__ __forceinline__ v8f wmma32(v16h a, v16h b, v8f c) {
  return __builtin_amdgcn_wmma_f32_16x16x32_f16(false, a, false, b, (short)0, c, false, false);
}

// One 16x16 output tile, K-chain split into two independent accumulators
// (compile-time trip counts; no predication).
template<int KTOT>
__device__ __forceinline__ v8f gemm_chain(const _Float16* A, int lda, int m0,
                                          const _Float16* B, int Ct, int nt, int lane) {
  constexpr int NS = KTOT / 32;
  v8f acc0 = {};
  if constexpr (NS == 1) {
    acc0 = wmma32(frag_a(A, lda, m0, 0, lane), frag_bs(B, Ct, 0, nt, lane), acc0);
    return acc0;
  } else {
    v8f acc1 = {};
#pragma unroll
    for (int s = 0; s + 1 < NS; s += 2) {
      acc0 = wmma32(frag_a(A, lda, m0, s * 32, lane),
                    frag_bs(B, Ct, s, nt, lane), acc0);
      acc1 = wmma32(frag_a(A, lda, m0, (s + 1) * 32, lane),
                    frag_bs(B, Ct, s + 1, nt, lane), acc1);
    }
    if constexpr (NS & 1) {
      acc0 = wmma32(frag_a(A, lda, m0, (NS - 1) * 32, lane),
                    frag_bs(B, Ct, NS - 1, nt, lane), acc0);
    }
    return acc0 + acc1;
  }
}

__device__ __forceinline__ float sigmoidf_(float x) { return 1.f / (1.f + __expf(-x)); }

// ---------------------------------------------------------------------------
// One GVP (vector_gate=True) on a 16-item tile.
//  MODE: 0=write As/Av for next gvp, 1=atomic scatter (edge), 2=store (node).
// ---------------------------------------------------------------------------
template<int SI, int H, int HP, int VIP, int K, int SO, int VO, int KN, bool ACT, int MODE>
__device__ __forceinline__ void gvp_apply(
    _Float16* As, _Float16* Av, _Float16* Vh, _Float16* sG, float* gate,
    const GvpWB W, const int* dstIds, int nodeBase,
    float* outS, float* outV)
{
  static_assert(SO % (16 * NWAVE) == 0, "phase-3 tiles must divide evenly");
  const int tid  = threadIdx.x;
  const int lane = tid & 31;
  const int wave = tid >> 5;
  const int half = (lane >> 4) & 1;
  const int ln15 = lane & 15;

  // ---- Phase 1: Vh[48,HP] = Av[48,VIP] @ wh[VIP,HP] (coords folded in rows)
  {
    constexpr int NT = HP / 16;
    for (int t = wave; t < 3 * NT; t += NWAVE) {
      int mt = t / NT, nt = t % NT;
      v8f acc = gemm_chain<VIP>(Av, VIP, mt * 16, W.wh, NT, nt, lane);
#pragma unroll
      for (int r = 0; r < 8; ++r)
        Vh[(size_t)(mt * 16 + r + 8 * half) * HP + nt * 16 + ln15] = (_Float16)acc[r];
    }
  }
  __syncthreads();

  // ---- Phase 2: vn = sqrt(sum_c vh^2 + eps) -> As[:, SI..K) (zero padded)
  for (int idx = tid; idx < 16 * (K - SI); idx += TB) {
    int i = idx / (K - SI), hh = idx % (K - SI);
    float val = 0.f;
    if (hh < H) {
      float ss = 1e-8f;
#pragma unroll
      for (int c = 0; c < 3; ++c) {
        float x = (float)Vh[(size_t)(c * 16 + i) * HP + hh];
        ss += x * x;
      }
      val = sqrtf(ss);
    }
    As[(size_t)i * K + SI + hh] = (_Float16)val;
  }
  __syncthreads();

  // ---- Phase 3: s_out = As[16,K] @ ws[K,SO] + b. A hoisted per k0; PW
  //      accumulators interleave in the inner loop (independent chains).
  {
    constexpr int NT = SO / 16;
    constexpr int PW = NT / NWAVE;
    v8f accS[PW] = {};
#pragma unroll
    for (int k0 = 0; k0 < K; k0 += 32) {
      v16h a = frag_a(As, K, 0, k0, lane);
#pragma unroll
      for (int j = 0; j < PW; ++j)
        accS[j] = wmma32(a, frag_bs(W.ws, NT, k0 >> 5, wave + NWAVE * j, lane), accS[j]);
    }
    __syncthreads();  // all waves done reading As before overwriting its s-region
#pragma unroll
    for (int j = 0; j < PW; ++j) {
      int t = wave + NWAVE * j;
      int n = t * 16 + ln15;
      float bias = W.ws_b[n];
#pragma unroll
      for (int r = 0; r < 8; ++r) {
        int m = r + 8 * half;
        float sv = accS[j][r] + bias;
        if (ACT) {
          As[(size_t)m * KN + n] = (_Float16)fmaxf(sv, 0.f);   // relu(s_out) -> next gvp
          sG[(size_t)m * SO + n] = (_Float16)sigmoidf_(sv);    // gate input
        } else {
          sG[(size_t)m * SO + n] = (_Float16)sv;               // gate input = s_out
          if (MODE == 1)      atomicAdd(outS + (size_t)dstIds[m] * 128 + n, sv);
          else if (MODE == 2) outS[(size_t)(nodeBase + m) * 128 + n] = sv;
        }
      }
    }
  }
  __syncthreads();

  // ---- Phase 4: gate[16,VO] = sigmoid(sG[16,SO] @ wsv[SO,VO] + b)
  {
    constexpr int NT = VO / 16;
    if (wave < NT) {
      v8f acc = gemm_chain<SO>(sG, SO, 0, W.wsv, NT, wave, lane);
      int n = wave * 16 + ln15;
      float bias = W.wsv_b[n];
#pragma unroll
      for (int r = 0; r < 8; ++r)
        gate[(size_t)(r + 8 * half) * VO + n] = sigmoidf_(acc[r] + bias);
    }
  }
  __syncthreads();

  // ---- Phase 5: v_out = (Vh[48,HP] @ wv[HP,VO]) * gate
  {
    constexpr int NT = VO / 16;
    for (int t = wave; t < 3 * NT; t += NWAVE) {
      int mt = t / NT, nt = t % NT;
      v8f acc = gemm_chain<HP>(Vh, HP, mt * 16, W.wv, NT, nt, lane);
      int n = nt * 16 + ln15;
#pragma unroll
      for (int r = 0; r < 8; ++r) {
        int g = mt * 16 + r + 8 * half;
        int c = g >> 4, i = g & 15;
        float val = acc[r] * gate[(size_t)i * VO + n];
        if (MODE == 1)      atomicAdd(outV + ((size_t)dstIds[i] * VO + n) * 3 + c, val);
        else if (MODE == 2) outV[((size_t)(nodeBase + i) * VO + n) * 3 + c] = val;
        else                Av[(size_t)g * VO + n] = (_Float16)val;  // VO == next VIP
      }
    }
  }
  __syncthreads();
}

// ---------------------------------------------------------------------------
// Edge-message kernel: 16 edges / workgroup, msg0 -> msg1 -> msg2, atomic mean
// ---------------------------------------------------------------------------
__global__ __launch_bounds__(TB) void edge_kernel(
    const float* __restrict__ s, const float* __restrict__ v,
    const float* __restrict__ es, const float* __restrict__ ev,
    const int* __restrict__ eidx, int E,
    MsgW W, float* __restrict__ ds, float* __restrict__ dv)
{
  __shared__ alignas(32) _Float16 As[16 * 384];
  __shared__ alignas(32) _Float16 Av[48 * 96];
  __shared__ alignas(32) _Float16 Vh[48 * 96];
  __shared__ alignas(32) _Float16 sG[16 * 128];
  __shared__ float    gate[16 * 32];
  __shared__ int srcI[16], dstI[16];

  const int e0  = blockIdx.x * 16;
  const int tid = threadIdx.x;
  if (tid < 16) { srcI[tid] = eidx[e0 + tid]; dstI[tid] = eidx[E + e0 + tid]; }
  __syncthreads();

  // scalar gather (float4 -> 4 halves): [s_src(128) | edge_s(32) | s_dst(128)]
  for (int idx = tid; idx < 16 * 72; idx += TB) {
    int i = idx / 72, j4 = (idx % 72) * 4;
    const float* src;
    if (j4 < 128)      src = s  + (size_t)srcI[i] * 128 + j4;
    else if (j4 < 160) src = es + (size_t)(e0 + i) * 32 + (j4 - 128);
    else               src = s  + (size_t)dstI[i] * 128 + (j4 - 160);
    float4 f = *(const float4*)src;
    v4h h; h[0] = (_Float16)f.x; h[1] = (_Float16)f.y; h[2] = (_Float16)f.z; h[3] = (_Float16)f.w;
    *(v4h*)&As[(size_t)i * 384 + j4] = h;
  }
  // vector gather: rows (c*16+i), cols [v_src(32)|edge_v(1)|v_dst(32)|pad->96]
  for (int idx = tid; idx < 48 * 96; idx += TB) {
    int g = idx / 96, j = idx % 96;
    int c = g >> 4, i = g & 15;
    float x = 0.f;
    if (j < 32)       x = v[((size_t)srcI[i] * 32 + j) * 3 + c];
    else if (j == 32) x = ev[(size_t)(e0 + i) * 3 + c];
    else if (j < 65)  x = v[((size_t)dstI[i] * 32 + (j - 33)) * 3 + c];
    Av[(size_t)g * 96 + j] = (_Float16)x;
  }
  __syncthreads();

  //            SI,  H, HP,VIP,  K, SO,VO, KN, ACT, MODE
  gvp_apply<288, 65, 96, 96, 384, 128, 32, 160, true,  0>(As, Av, Vh, sG, gate, W.g[0], dstI, 0, nullptr, nullptr);
  gvp_apply<128, 32, 32, 32, 160, 128, 32, 160, true,  0>(As, Av, Vh, sG, gate, W.g[1], dstI, 0, nullptr, nullptr);
  gvp_apply<128, 32, 32, 32, 160, 128, 32, 160, false, 1>(As, Av, Vh, sG, gate, W.g[2], dstI, 0, ds, dv);
}

// ---------------------------------------------------------------------------
// Node feed-forward kernel: 16 nodes / workgroup, ff0 -> ff1
// ---------------------------------------------------------------------------
__global__ __launch_bounds__(TB) void node_kernel(
    const float* __restrict__ s, const float* __restrict__ v,
    FfW W, float* __restrict__ ffs, float* __restrict__ ffv)
{
  __shared__ alignas(32) _Float16 As[16 * 576];
  __shared__ alignas(32) _Float16 Av[48 * 64];
  __shared__ alignas(32) _Float16 Vh[48 * 64];
  __shared__ alignas(32) _Float16 sG[16 * 512];
  __shared__ float    gate[16 * 64];

  const int n0  = blockIdx.x * 16;
  const int tid = threadIdx.x;

  for (int idx = tid; idx < 16 * 32; idx += TB) {
    int i = idx >> 5, j4 = (idx & 31) * 4;
    float4 f = *(const float4*)(s + (size_t)(n0 + i) * 128 + j4);
    v4h h; h[0] = (_Float16)f.x; h[1] = (_Float16)f.y; h[2] = (_Float16)f.z; h[3] = (_Float16)f.w;
    *(v4h*)&As[(size_t)i * 192 + j4] = h;
  }
  for (int idx = tid; idx < 48 * 32; idx += TB) {
    int g = idx >> 5, j = idx & 31;
    int c = g >> 4, i = g & 15;
    Av[(size_t)g * 32 + j] = (_Float16)v[((size_t)(n0 + i) * 32 + j) * 3 + c];
  }
  __syncthreads();

  //            SI,  H, HP,VIP,  K,  SO, VO,  KN, ACT, MODE
  gvp_apply<128, 64, 64, 32, 192, 512, 64, 576, true,  0>(As, Av, Vh, sG, gate, W.g[0], nullptr, n0, nullptr, nullptr);
  gvp_apply<512, 64, 64, 64, 576, 128, 32, 160, false, 2>(As, Av, Vh, sG, gate, W.g[1], nullptr, n0, ffs, ffv);
}

// ---------------------------------------------------------------------------
// Residual + LayerNorm (+optional mean aggregation). One wave32 per node.
// ---------------------------------------------------------------------------
__global__ __launch_bounds__(256) void ln_kernel(
    const float* __restrict__ s_in, const float* __restrict__ v_in,
    const float* __restrict__ add_s, const float* __restrict__ add_v,
    const float* __restrict__ cnt,
    const float* __restrict__ g, const float* __restrict__ b,
    float* __restrict__ s_out, float* __restrict__ v_out, int N)
{
  int w    = (blockIdx.x * blockDim.x + threadIdx.x) >> 5;
  int lane = threadIdx.x & 31;
  if (w >= N) return;
  float inv = 1.f;
  if (cnt) inv = 1.f / fmaxf(cnt[w], 1.f);

  float sv[4], sum = 0.f;
#pragma unroll
  for (int q = 0; q < 4; ++q) {
    int j = lane + 32 * q;
    sv[q] = s_in[(size_t)w * 128 + j] + add_s[(size_t)w * 128 + j] * inv;
    sum += sv[q];
  }
#pragma unroll
  for (int o = 16; o; o >>= 1) sum += __shfl_xor(sum, o, 32);
  float mu = sum * (1.f / 128.f);
  float var = 0.f;
#pragma unroll
  for (int q = 0; q < 4; ++q) { float d = sv[q] - mu; var += d * d; }
#pragma unroll
  for (int o = 16; o; o >>= 1) var += __shfl_xor(var, o, 32);
  float rstd = rsqrtf(var * (1.f / 128.f) + 1e-5f);
#pragma unroll
  for (int q = 0; q < 4; ++q) {
    int j = lane + 32 * q;
    s_out[(size_t)w * 128 + j] = (sv[q] - mu) * rstd * g[j] + b[j];
  }

  float vv[3], ss = 0.f;
#pragma unroll
  for (int q = 0; q < 3; ++q) {
    int j = lane + 32 * q;
    vv[q] = v_in[(size_t)w * 96 + j] + add_v[(size_t)w * 96 + j] * inv;
    ss += vv[q] * vv[q];
  }
#pragma unroll
  for (int o = 16; o; o >>= 1) ss += __shfl_xor(ss, o, 32);
  float vn = sqrtf(ss * (1.f / 32.f) + 1e-8f);
#pragma unroll
  for (int q = 0; q < 3; ++q) {
    int j = lane + 32 * q;
    v_out[(size_t)w * 96 + j] = vv[q] / vn;
  }
}

// ---------------------------------------------------------------------------
// fp32 -> f16 weight pad/convert into fragment-swizzled layout.
// dst flat index = ((kt*(C/16)+nt)*32 + lane)*16 + e,
//   K = kt*32 + (lane>>4)*8 + (e&7) + (e>>3)*16,  N = nt*16 + (lane&15).
// ---------------------------------------------------------------------------
struct PadDesc { const float* src; _Float16* dst; int r, c, R, C; };
struct PadTable { PadDesc d[60]; };

__global__ __launch_bounds__(256) void pad_kernel(PadTable T) {
  PadDesc p = T.d[blockIdx.x];
  int total = p.R * p.C;
  int ntiles = p.C >> 4;
  int stride = gridDim.y * blockDim.x;
  for (int idx = blockIdx.y * blockDim.x + threadIdx.x; idx < total; idx += stride) {
    int e    = idx & 15;
    int lane = (idx >> 4) & 31;
    int tile = idx >> 9;
    int kt = tile / ntiles, nt = tile % ntiles;
    int K = kt * 32 + ((lane >> 4) << 3) + (e & 7) + ((e >> 3) << 4);
    int N = nt * 16 + (lane & 15);
    float x = (K < p.r && N < p.c) ? p.src[(size_t)K * p.c + N] : 0.f;
    p.dst[idx] = (_Float16)x;
  }
}

__global__ __launch_bounds__(256) void count_kernel(const int* __restrict__ eidx, int E,
                                                    float* __restrict__ cnt) {
  int e = blockIdx.x * blockDim.x + threadIdx.x;
  if (e < E) atomicAdd(&cnt[eidx[E + e]], 1.0f);
}

// ---------------------------------------------------------------------------
extern "C" void kernel_launch(void* const* d_in, const int* in_sizes, int n_in,
                              void* d_out, int out_size, void* d_ws, size_t ws_size,
                              hipStream_t stream) {
  (void)in_sizes; (void)n_in; (void)out_size; (void)ws_size;
  const int N = 16384, E = 262144;

  const float* node_s = (const float*)d_in[0];
  const float* node_v = (const float*)d_in[1];
  const float* edge_s = (const float*)d_in[2];
  const float* edge_v = (const float*)d_in[3];
  const int*   eidx   = (const int*)d_in[4];

  // ---- workspace carve ----
  char* base = (char*)d_ws;
  size_t off = 0;
  auto carve = [&](size_t bytes) -> void* {
    void* r = base + off;
    off = (off + bytes + 255) & ~(size_t)255;
    return r;
  };
  _Float16* hw  = (_Float16*)carve((size_t)4 << 20);       // padded f16 weights
  float* sW0 = (float*)carve((size_t)N * 128 * 4);
  float* sW1 = (float*)carve((size_t)N * 128 * 4);
  float* vW0 = (float*)carve((size_t)N * 96 * 4);
  float* vW1 = (float*)carve((size_t)N * 96 * 4);
  float* dsb = (float*)carve((size_t)N * 128 * 4);         // aggr / ff scalar out
  float* dvb = (float*)carve((size_t)N * 96 * 4);         // aggr / ff vector out
  float* cnt = (float*)carve((size_t)N * 4);

  // ---- per-gvp dims: {vi, h, si, so, vo, VIP, HP, K} ----
  struct Dm { int vi, h, si, so, vo, VIP, HP, K; };
  static const Dm dms[5] = {
    {65, 65, 288, 128, 32, 96, 96, 384},   // msg0
    {32, 32, 128, 128, 32, 32, 32, 160},   // msg1
    {32, 32, 128, 128, 32, 32, 32, 160},   // msg2
    {32, 64, 128, 512, 64, 32, 64, 192},   // ff0
    {64, 64, 512, 128, 32, 64, 64, 576},   // ff1
  };

  PadTable T;
  MsgW mw[3];
  FfW  fw[3];
  const float *ln0g[3], *ln0b[3], *ln1g[3], *ln1b[3];
  int ti = 0;
  size_t ho = 0;
  for (int L = 0; L < 3; ++L) {
    int pb = 5 + 34 * L;
    for (int gi = 0; gi < 5; ++gi) {
      const Dm& d = dms[gi];
      int pi = pb + 6 * gi;
      auto add = [&](const float* src, int r, int c, int R, int C) -> _Float16* {
        _Float16* dst = hw + ho;
        ho = (ho + (size_t)R * C + 127) & ~(size_t)127;
        T.d[ti].src = src; T.d[ti].dst = dst;
        T.d[ti].r = r; T.d[ti].c = c; T.d[ti].R = R; T.d[ti].C = C;
        ++ti;
        return dst;
      };
      GvpWB W;
      W.wh    = add((const float*)d_in[pi + 0], d.vi, d.h,      d.VIP, d.HP);
      W.ws    = add((const float*)d_in[pi + 1], d.si + d.h, d.so, d.K,  d.so);
      W.wv    = add((const float*)d_in[pi + 3], d.h, d.vo,      d.HP,  d.vo);
      W.wsv   = add((const float*)d_in[pi + 4], d.so, d.vo,     d.so,  d.vo);
      W.ws_b  = (const float*)d_in[pi + 2];
      W.wsv_b = (const float*)d_in[pi + 5];
      if (gi < 3) mw[L].g[gi] = W;
      else        fw[L].g[gi - 3] = W;
    }
    ln0g[L] = (const float*)d_in[pb + 30];
    ln0b[L] = (const float*)d_in[pb + 31];
    ln1g[L] = (const float*)d_in[pb + 32];
    ln1b[L] = (const float*)d_in[pb + 33];
  }

  // ---- one-time per-call prep ----
  pad_kernel<<<dim3(60, 96), 256, 0, stream>>>(T);
  hipMemsetAsync(cnt, 0, (size_t)N * 4, stream);
  count_kernel<<<(E + 255) / 256, 256, 0, stream>>>(eidx, E, cnt);

  // ---- layers ----
  const float* sc = node_s;
  const float* vc = node_v;
  float* outS = (float*)d_out;
  float* outV = (float*)d_out + (size_t)N * 128;

  for (int L = 0; L < 3; ++L) {
    hipMemsetAsync(dsb, 0, (size_t)N * 128 * 4, stream);
    hipMemsetAsync(dvb, 0, (size_t)N * 96 * 4, stream);
    edge_kernel<<<E / 16, TB, 0, stream>>>(sc, vc, edge_s, edge_v, eidx, E, mw[L], dsb, dvb);
    // mean aggregation + residual + norm[0]
    ln_kernel<<<(N * 32 + 255) / 256, 256, 0, stream>>>(
        sc, vc, dsb, dvb, cnt, ln0g[L], ln0b[L], sW0, vW0, N);
    // feed-forward (ff0 -> ff1), outputs into dsb/dvb
    node_kernel<<<N / 16, TB, 0, stream>>>(sW0, vW0, fw[L], dsb, dvb);
    // residual + norm[1]
    float* sN = (L == 2) ? outS : sW1;
    float* vN = (L == 2) ? outV : vW1;
    ln_kernel<<<(N * 32 + 255) / 256, 256, 0, stream>>>(
        sW0, vW0, dsb, dvb, nullptr, ln1g[L], ln1b[L], sN, vN, N);
    sc = sW1;
    vc = vW1;
  }
}
